// Attention_695784702572
// MI455X (gfx1250) — compile-verified, matrix operand order checked by reading, not run on hardware
//
#include <hip/hip_runtime.h>
#include <hip/hip_bf16.h>
#include <math.h>

// ---------------------------------------------------------------------------
// Types for CDNA5 WMMA (wave32)
// ---------------------------------------------------------------------------
typedef __bf16 v16bf __attribute__((ext_vector_type(16)));
typedef __bf16 v8bf  __attribute__((ext_vector_type(8)));
typedef float  v8f   __attribute__((ext_vector_type(8)));
typedef int    ivec4 __attribute__((ext_vector_type(4)));

#define AS1 __attribute__((address_space(1)))
#define AS3 __attribute__((address_space(3)))

// Optional gfx1250 async global->LDS copy path (safe-guarded).
#if defined(__has_builtin)
#if __has_builtin(__builtin_amdgcn_global_load_async_to_lds_b128) && \
    __has_builtin(__builtin_amdgcn_s_wait_asynccnt)
#define USE_ASYNC_LDS 1
#endif
#endif

// ---------------------------------------------------------------------------
// Helpers
// ---------------------------------------------------------------------------
__device__ __forceinline__ unsigned short f2bf(float f) {
    unsigned int u = __builtin_bit_cast(unsigned int, f);
    return (unsigned short)((u + 0x7fffu + ((u >> 16) & 1u)) >> 16);  // RNE
}
__device__ __forceinline__ float bf2f(unsigned short u) {
    return __builtin_bit_cast(float, ((unsigned int)u) << 16);
}
__device__ __forceinline__ v8f wmma_bf16(v16bf a, v16bf b, v8f c) {
    return __builtin_amdgcn_wmma_f32_16x16x32_bf16(
        false, a, false, b, (short)0, c, false, false);
}
// Fragment load for one 32-wide K window laid out contiguously at `base`
// (16 lanes own rows/cols; lane-half hf picks K {hf*8..} and {16+hf*8..}).
// Two 16-byte loads -> ds_load_b128 / global_load_b128.
__device__ __forceinline__ v16bf load_frag32(const unsigned short* base, int hf) {
    v8bf lo = *(const v8bf*)(base + hf * 8);
    v8bf hi = *(const v8bf*)(base + 16 + hf * 8);
    return __builtin_shufflevector(lo, hi, 0, 1, 2, 3, 4, 5, 6, 7,
                                   8, 9, 10, 11, 12, 13, 14, 15);
}

// ---------------------------------------------------------------------------
// Constants
// ---------------------------------------------------------------------------
#define BATCH 2
#define HEADS 8
#define DH    64
#define MTOK  2048             // l*n
#define ROWS  (BATCH * MTOK)   // 4096
#define DIM   512
#define INNER 512
#define NQKV  1536

// ---------------------------------------------------------------------------
// Kernel 1: f32 -> bf16 weight conversion
// ---------------------------------------------------------------------------
__global__ void cvt_bf16_kernel(const float* __restrict__ in,
                                unsigned short* __restrict__ out, int n) {
    int i = blockIdx.x * blockDim.x + threadIdx.x;
    if (i < n) out[i] = f2bf(in[i]);
}

// ---------------------------------------------------------------------------
// Kernel 2: xyzw = xyz @ w_sp    ([4096,3] x [3,64] -> [4096,64] f32)
// ---------------------------------------------------------------------------
__global__ void xyzw_kernel(const float* __restrict__ xyzs,
                            const float* __restrict__ wsp,
                            float* __restrict__ xyzw) {
    int idx = blockIdx.x * blockDim.x + threadIdx.x;
    if (idx >= ROWS * DH) return;
    int i = idx >> 6, d = idx & 63;
    const float* p = xyzs + (size_t)i * 3;
    xyzw[idx] = p[0] * wsp[d] + p[1] * wsp[64 + d] + p[2] * wsp[128 + d];
}

// ---------------------------------------------------------------------------
// Kernel 3: LayerNorm over dim=512, output bf16. One block per row.
// ---------------------------------------------------------------------------
__global__ __launch_bounds__(256) void layernorm_kernel(
    const float* __restrict__ feat, const float* __restrict__ gamma,
    const float* __restrict__ beta, unsigned short* __restrict__ normed) {
    __shared__ float red[256];
    int t = threadIdx.x;
    const float* row = feat + (size_t)blockIdx.x * DIM;
    float x0 = row[t], x1 = row[t + 256];
    red[t] = x0 + x1;
    __syncthreads();
    for (int o = 128; o > 0; o >>= 1) {
        if (t < o) red[t] += red[t + o];
        __syncthreads();
    }
    float mean = red[0] * (1.0f / DIM);
    __syncthreads();
    red[t] = x0 * x0 + x1 * x1;
    __syncthreads();
    for (int o = 128; o > 0; o >>= 1) {
        if (t < o) red[t] += red[t + o];
        __syncthreads();
    }
    float var  = red[0] * (1.0f / DIM) - mean * mean;
    float rstd = rsqrtf(var + 1e-5f);
    unsigned short* orow = normed + (size_t)blockIdx.x * DIM;
    orow[t]       = f2bf((x0 - mean) * rstd * gamma[t]       + beta[t]);
    orow[t + 256] = f2bf((x1 - mean) * rstd * gamma[t + 256] + beta[t + 256]);
}

// ---------------------------------------------------------------------------
// Shared WMMA GEMM mainloop: C[128x128] block tile, K=512, bf16 in, f32 acc.
// 256 threads = 8 waves arranged 4(M) x 2(N); each wave computes 32x64 as a
// 2x4 grid of 16x16 WMMA accumulators (8 WMMA per 6 fragment loads).
// A tile [128][32]; B tile stored TRANSPOSED [128 cols][32 k]; every
// fragment is 2x ds_load_b128.
// ---------------------------------------------------------------------------
__device__ __forceinline__ void gemm_tile_k512(
    const unsigned short* __restrict__ Ag,  // [M][512] bf16 row-major
    const unsigned short* __restrict__ Bg,  // [512][N] bf16 row-major
    int N, int bm, int bn,
    unsigned short* As /*[128*32], 16B aligned*/,
    unsigned short* BsT /*[128*32], 16B aligned*/, v8f c[2][4]) {
    unsigned int* Asu = (unsigned int*)As;
    const unsigned int* Agu = (const unsigned int*)Ag;
    const unsigned int* Bgu = (const unsigned int*)Bg;
    int tid = threadIdx.x, lane = tid & 31, wid = tid >> 5;
    int wm = wid & 3, wn = wid >> 2, hf = lane >> 4, cl = lane & 15;
    int Nu = N >> 1;
    for (int k0 = 0; k0 < 512; k0 += 32) {
        if (k0 + 32 < 512) {  // warm L2 for next tiles (global_prefetch_b8)
            __builtin_prefetch(&Agu[(size_t)bm * 256 + ((k0 + 32) >> 1)], 0, 1);
            __builtin_prefetch(&Bgu[(size_t)(k0 + 32) * Nu + (bn >> 1)], 0, 1);
        }
#pragma unroll
        for (int i = 0; i < 8; ++i) {  // A tile: 128x32 bf16 = 2048 dwords
            int lin = tid + i * 256, r = lin >> 4, cu = lin & 15;
            Asu[lin] = Agu[(size_t)(bm + r) * 256 + (k0 >> 1) + cu];
        }
#pragma unroll
        for (int i = 0; i < 8; ++i) {  // B tile transposed: [128][32]
            int lin = tid + i * 256, r = lin >> 6, cu = lin & 63;
            unsigned int w = Bgu[(size_t)(k0 + r) * Nu + (bn >> 1) + cu];
            int n0 = cu * 2;
            BsT[n0 * 32 + r]       = (unsigned short)(w & 0xffffu);
            BsT[(n0 + 1) * 32 + r] = (unsigned short)(w >> 16);
        }
        __syncthreads();
        v16bf a[2], b[4];
#pragma unroll
        for (int sm = 0; sm < 2; ++sm)
            a[sm] = load_frag32(As + (wm * 32 + sm * 16 + cl) * 32, hf);
#pragma unroll
        for (int sn = 0; sn < 4; ++sn)
            b[sn] = load_frag32(BsT + (wn * 64 + sn * 16 + cl) * 32, hf);
#pragma unroll
        for (int sm = 0; sm < 2; ++sm)
#pragma unroll
            for (int sn = 0; sn < 4; ++sn)
                c[sm][sn] = wmma_bf16(a[sm], b[sn], c[sm][sn]);
        __syncthreads();
    }
}

// ---------------------------------------------------------------------------
// Kernel 4: QKV GEMM. normed[4096,512] @ w_qkv[512,1536].
// ---------------------------------------------------------------------------
__global__ __launch_bounds__(256) void gemm_qkv_kernel(
    const unsigned short* __restrict__ normed,
    const unsigned short* __restrict__ wqkv,
    const float* __restrict__ xyzw,
    unsigned short* __restrict__ qb, unsigned short* __restrict__ kb,
    unsigned short* __restrict__ vpb) {
    __shared__ __attribute__((aligned(16))) unsigned short As[128 * 32];
    __shared__ __attribute__((aligned(16))) unsigned short BsT[128 * 32];
    const int nbt = NQKV / 128;  // 12
    int bm = (blockIdx.x / nbt) * 128, bn = (blockIdx.x % nbt) * 128;
    v8f c[2][4] = {};
    gemm_tile_k512(normed, wqkv, NQKV, bm, bn, As, BsT, c);

    int tid = threadIdx.x, lane = tid & 31, wid = tid >> 5;
    int wm = wid & 3, wn = wid >> 2, hf = lane >> 4, cl = lane & 15;
#pragma unroll
    for (int sm = 0; sm < 2; ++sm)
#pragma unroll
        for (int sn = 0; sn < 4; ++sn)
#pragma unroll
            for (int r = 0; r < 8; ++r) {
                int grow = bm + wm * 32 + sm * 16 + r + 8 * hf;
                int gcol = bn + wn * 64 + sn * 16 + cl;
                float v = c[sm][sn][r];
                int chunk = gcol / INNER, rest = gcol % INNER;
                int h = rest >> 6, d = rest & 63;
                int bt = grow >> 11, mm = grow & 2047;
                size_t dst = (((size_t)bt * HEADS + h) * MTOK + mm) * DH + d;
                if (chunk == 0)      qb[dst]  = f2bf(v * 0.125f);  // DH^-0.5
                else if (chunk == 1) kb[dst]  = f2bf(v);
                else                 vpb[dst] = f2bf(v + xyzw[((size_t)bt * MTOK + mm) * DH + d]);
            }
}

// ---------------------------------------------------------------------------
// Kernel 5: flash attention per (b,h). Block = 128 threads = 4 waves.
// K tile copied via async global->LDS (row-major, used directly);
// V tile stored transposed so PV B-fragments are contiguous b128 loads.
// ---------------------------------------------------------------------------
__global__ __launch_bounds__(128) void attn_kernel(
    const unsigned short* __restrict__ qb, const unsigned short* __restrict__ kb,
    const unsigned short* __restrict__ vpb, const float* __restrict__ xyzw,
    unsigned short* __restrict__ attn_cat) {
    __shared__ __attribute__((aligned(16))) unsigned short kl[32 * 64];   // [key][d]
    __shared__ __attribute__((aligned(16))) unsigned short vlT[64 * 32];  // [d][key]
    __shared__ __attribute__((aligned(16))) unsigned short pls[4 * 16 * 32];

    int tid = threadIdx.x, wid = tid >> 5, lane = tid & 31;
    int hf = lane >> 4, cl = lane & 15;
    int bh = blockIdx.x >> 5, qt = blockIdx.x & 31;
    int b = bh >> 3, h = bh & 7;

    const unsigned short* qg = qb  + (size_t)bh * MTOK * DH;
    const unsigned short* kg = kb  + (size_t)bh * MTOK * DH;
    const unsigned short* vg = vpb + (size_t)bh * MTOK * DH;

    // Preload Q fragments (2x b128 each, straight from global).
    int qrow = qt * 64 + wid * 16 + cl;
    v16bf aq[2];
#pragma unroll
    for (int ds = 0; ds < 2; ++ds)
        aq[ds] = load_frag32(qg + (size_t)qrow * DH + ds * 32, hf);

    v8f o[4] = {};
    float rmax[8], rsum[8];
#pragma unroll
    for (int r = 0; r < 8; ++r) { rmax[r] = -3.0e38f; rsum[r] = 0.0f; }
    unsigned short* myP = pls + wid * 512;

    for (int kc = 0; kc < MTOK; kc += 32) {
        // ---- K tile: contiguous 4KB copy ----
#ifdef USE_ASYNC_LDS
        {
            const char* gsrc = (const char*)kg + (size_t)kc * DH * 2 + tid * 16;
            char* ldst = (char*)kl + tid * 16;
#pragma unroll
            for (int i = 0; i < 2; ++i) {
                __builtin_amdgcn_global_load_async_to_lds_b128(
                    (AS1 ivec4*)(gsrc + i * 2048),
                    (AS3 ivec4*)(ldst + i * 2048),
                    0, 0);
            }
        }
#else
        {
            const unsigned int* ksrc = (const unsigned int*)kg + (size_t)kc * 32;
            unsigned int* klu = (unsigned int*)kl;
#pragma unroll
            for (int i = 0; i < 8; ++i) klu[tid + i * 128] = ksrc[tid + i * 128];
        }
#endif
        // ---- V tile: transposed store [d][key] ----
        {
            const unsigned int* vsrc = (const unsigned int*)vg + (size_t)kc * 32;
#pragma unroll
            for (int i = 0; i < 8; ++i) {
                int lin = tid + i * 128, key = lin >> 5, du = lin & 31;
                unsigned int w = vsrc[lin];
                int d0 = du * 2;
                vlT[d0 * 32 + key]       = (unsigned short)(w & 0xffffu);
                vlT[(d0 + 1) * 32 + key] = (unsigned short)(w >> 16);
            }
        }
#ifdef USE_ASYNC_LDS
        __builtin_amdgcn_s_wait_asynccnt(0);
#endif
        __syncthreads();

        // ---- S = Q K^T ----
        v8f st[2];
#pragma unroll
        for (int t = 0; t < 2; ++t) {
            v8f s = {};
#pragma unroll
            for (int ds = 0; ds < 2; ++ds) {
                v16bf bk = load_frag32(kl + (t * 16 + cl) * DH + ds * 32, hf);
                s = wmma_bf16(aq[ds], bk, s);
            }
            st[t] = s;
        }

        // ---- online softmax over this 32-key chunk ----
#pragma unroll
        for (int r = 0; r < 8; ++r) {
            float s0 = st[0][r], s1 = st[1][r];
            float mx = fmaxf(s0, s1);
#pragma unroll
            for (int m = 1; m < 16; m <<= 1) mx = fmaxf(mx, __shfl_xor(mx, m, 32));
            float mnew = fmaxf(rmax[r], mx);
            float corr = __expf(rmax[r] - mnew);
            float p0 = __expf(s0 - mnew), p1 = __expf(s1 - mnew);
            float ps = p0 + p1;
#pragma unroll
            for (int m = 1; m < 16; m <<= 1) ps += __shfl_xor(ps, m, 32);
            rsum[r] = rsum[r] * corr + ps;
            rmax[r] = mnew;
#pragma unroll
            for (int nt = 0; nt < 4; ++nt) o[nt][r] *= corr;
            int mrow = r + 8 * hf;
            myP[mrow * 32 + cl]      = f2bf(p0);
            myP[mrow * 32 + 16 + cl] = f2bf(p1);
        }

        // ---- O += P @ Vp ----
        v16bf ap = load_frag32(myP + cl * 32, hf);
#pragma unroll
        for (int nt = 0; nt < 4; ++nt) {
            v16bf bv = load_frag32(vlT + (nt * 16 + cl) * 32, hf);
            o[nt] = wmma_bf16(ap, bv, o[nt]);
        }
        __syncthreads();
    }

    // Epilogue: normalize, subtract xyzw_i, store bf16 into [b][m][h*64+d].
#pragma unroll
    for (int r = 0; r < 8; ++r) {
        float inv = 1.0f / rsum[r];
        int m = qt * 64 + wid * 16 + r + 8 * hf;
        size_t gm = (size_t)b * MTOK + m;
#pragma unroll
        for (int nt = 0; nt < 4; ++nt) {
            int d = nt * 16 + cl;
            float val = o[nt][r] * inv - xyzw[gm * DH + d];
            attn_cat[gm * DIM + h * DH + d] = f2bf(val);
        }
    }
}

// ---------------------------------------------------------------------------
// Kernel 6: output GEMM attn_cat[4096,512] @ w_out[512,512]; + b_out,
// exact GELU, + residual features; f32 out.
// ---------------------------------------------------------------------------
__global__ __launch_bounds__(256) void gemm_out_kernel(
    const unsigned short* __restrict__ attn_cat,
    const unsigned short* __restrict__ wout,
    const float* __restrict__ b_out, const float* __restrict__ feat,
    float* __restrict__ out) {
    __shared__ __attribute__((aligned(16))) unsigned short As[128 * 32];
    __shared__ __attribute__((aligned(16))) unsigned short BsT[128 * 32];
    const int nbt = DIM / 128;  // 4
    int bm = (blockIdx.x / nbt) * 128, bn = (blockIdx.x % nbt) * 128;
    v8f c[2][4] = {};
    gemm_tile_k512(attn_cat, wout, DIM, bm, bn, As, BsT, c);

    int tid = threadIdx.x, lane = tid & 31, wid = tid >> 5;
    int wm = wid & 3, wn = wid >> 2, hf = lane >> 4, cl = lane & 15;
#pragma unroll
    for (int sm = 0; sm < 2; ++sm)
#pragma unroll
        for (int sn = 0; sn < 4; ++sn)
#pragma unroll
            for (int r = 0; r < 8; ++r) {
                int grow = bm + wm * 32 + sm * 16 + r + 8 * hf;
                int gcol = bn + wn * 64 + sn * 16 + cl;
                float x = c[sm][sn][r] + b_out[gcol];
                float g = 0.5f * x * (1.0f + erff(x * 0.70710678118654752f));
                size_t idx = (size_t)grow * DIM + gcol;
                out[idx] = g + feat[idx];
            }
}

// ---------------------------------------------------------------------------
// Launch
// ---------------------------------------------------------------------------
extern "C" void kernel_launch(void* const* d_in, const int* in_sizes, int n_in,
                              void* d_out, int out_size, void* d_ws, size_t ws_size,
                              hipStream_t stream) {
    (void)in_sizes; (void)n_in; (void)out_size; (void)ws_size;
    const float* xyzs  = (const float*)d_in[0];
    const float* feat  = (const float*)d_in[1];
    const float* gamma = (const float*)d_in[2];
    const float* beta  = (const float*)d_in[3];
    const float* w_qkv = (const float*)d_in[4];
    const float* w_sp  = (const float*)d_in[5];
    const float* w_out = (const float*)d_in[6];
    const float* b_out = (const float*)d_in[7];
    float* out = (float*)d_out;

    size_t off = 0;
    auto alloc = [&](size_t bytes) {
        void* p = (char*)d_ws + off;
        off += (bytes + 255) & ~(size_t)255;
        return p;
    };
    unsigned short* normed = (unsigned short*)alloc((size_t)ROWS * DIM * 2);
    unsigned short* wqkv_b = (unsigned short*)alloc((size_t)DIM * NQKV * 2);
    unsigned short* wout_b = (unsigned short*)alloc((size_t)DIM * DIM * 2);
    float*          xyzw   = (float*)alloc((size_t)ROWS * DH * 4);
    unsigned short* qb     = (unsigned short*)alloc((size_t)ROWS * INNER * 2);
    unsigned short* kbuf   = (unsigned short*)alloc((size_t)ROWS * INNER * 2);
    unsigned short* vpb    = (unsigned short*)alloc((size_t)ROWS * INNER * 2);
    unsigned short* attn_c = (unsigned short*)alloc((size_t)ROWS * DIM * 2);

    cvt_bf16_kernel<<<(DIM * NQKV + 255) / 256, 256, 0, stream>>>(w_qkv, wqkv_b, DIM * NQKV);
    cvt_bf16_kernel<<<(DIM * DIM + 255) / 256, 256, 0, stream>>>(w_out, wout_b, DIM * DIM);
    xyzw_kernel<<<(ROWS * DH + 255) / 256, 256, 0, stream>>>(xyzs, w_sp, xyzw);
    layernorm_kernel<<<ROWS, 256, 0, stream>>>(feat, gamma, beta, normed);
    gemm_qkv_kernel<<<(ROWS / 128) * (NQKV / 128), 256, 0, stream>>>(
        normed, wqkv_b, xyzw, qb, kbuf, vpb);
    attn_kernel<<<BATCH * HEADS * (MTOK / 64), 128, 0, stream>>>(
        qb, kbuf, vpb, xyzw, attn_c);
    gemm_out_kernel<<<(ROWS / 128) * (DIM / 128), 256, 0, stream>>>(
        attn_c, wout_b, b_out, feat, out);
}
